// BiMixtureOfAdapters_14972255994051
// MI455X (gfx1250) — matrix operand order repaired and checked by reading, not compiled, not run on hardware
//
#include <hip/hip_runtime.h>
#include <hip/hip_bf16.h>

typedef __bf16  v16bf __attribute__((ext_vector_type(16)));
typedef float   v8f   __attribute__((ext_vector_type(8)));

#define BB 8
#define BN 2048
#define DIMC 1024
#define DRED 256
#define NEXP 4
#define DHID 32
#define MTOK (BB*BN)   // 16384

__device__ __forceinline__ __bf16 f2bf(float f) {
  unsigned u = __builtin_bit_cast(unsigned, f);
  u += 0x7FFFu + ((u >> 16) & 1u);               // round-to-nearest-even
  unsigned short h = (unsigned short)(u >> 16);
  return __builtin_bit_cast(__bf16, h);
}

// A-matrix (16x32 bf16) element->K mapping (contiguous, matches B):
// lanes 0-15 hold K in {0..15}; lanes 16-31 hold K in {16..31}
__device__ __forceinline__ int a_k_of_e(int e, int half) {
  int v = e >> 1, p = e & 1;
  (void)v; (void)p;
  return e + (half << 4);
}

// ---------------------------------------------------------------------------
// k1: LN1(concat(x,t)) -> GEMM(w_red) [WMMA bf16] -> LN2 -> gate logits/top2
// block = 256 thr (8 waves), 32 rows x 256 cols per block
// ---------------------------------------------------------------------------
__global__ __launch_bounds__(256)
void k1_ln_reduce(const float* __restrict__ x, const float* __restrict__ t,
                  const float* __restrict__ ln1_g, const float* __restrict__ ln1_b,
                  const float* __restrict__ w_red,
                  const float* __restrict__ ln2_g, const float* __restrict__ ln2_b,
                  const float* __restrict__ w_gate, const int* __restrict__ taskp,
                  float* __restrict__ yf, float* __restrict__ gates)
{
  __shared__ float rowMean[32], rowRstd[32];
  __shared__ float red[256], red2[256];
  __shared__ v16bf aF[2*32];        // 2 row-tiles, fragment layout
  __shared__ v16bf bF[16*32];       // 16 col-tiles, fragment layout
  __shared__ float yBuf[32*256];
  __shared__ float lgs[32*4];

  const int tid  = threadIdx.x;
  const int lane = tid & 31;
  const int w    = tid >> 5;
  const int gm0  = blockIdx.x * 32;

  // ---- phase 0: LN1 row stats over 2048 features ----
  {
    int r = tid >> 3, s = tid & 7;
    const float* xr = x + (size_t)(gm0 + r) * DIMC;
    const float* tr = t + (size_t)(gm0 + r) * DIMC;
    float sm = 0.f, sq = 0.f;
    for (int i = 0; i < 256; ++i) {
      int f = s + 8 * i;
      float v = (f < DIMC) ? xr[f] : tr[f - DIMC];
      sm += v; sq += v * v;
    }
    red[tid] = sm; red2[tid] = sq;
  }
  __syncthreads();
  if (tid < 32) {
    float sm = 0.f, sq = 0.f;
    for (int j = 0; j < 8; ++j) { sm += red[tid*8+j]; sq += red2[tid*8+j]; }
    float mean = sm * (1.f/2048.f);
    float var  = sq * (1.f/2048.f) - mean*mean;
    rowMean[tid] = mean;
    rowRstd[tid] = rsqrtf(var + 1e-5f);
  }
  __syncthreads();

  // ---- phase 1: K loop, WMMA ----
  const int rt = w >> 2;   // row tile (0/1)
  const int cg = w & 3;    // column group: tiles cg*4 .. cg*4+3
  v8f acc[4];
  #pragma unroll
  for (int j = 0; j < 4; ++j)
    #pragma unroll
    for (int r8 = 0; r8 < 8; ++r8) acc[j][r8] = 0.f;

  for (int kt = 0; kt < 64; ++kt) {
    const int K0 = kt * 32;
    __syncthreads();
    // stage A: normalize 32 rows x 32 k, write in fragment order
    #pragma unroll
    for (int ii = 0; ii < 4; ++ii) {
      int lin = tid + 256*ii;
      int m = lin >> 5, k = lin & 31;
      int kg = K0 + k;
      float v = (kg < DIMC) ? x[(size_t)(gm0+m)*DIMC + kg]
                            : t[(size_t)(gm0+m)*DIMC + (kg - DIMC)];
      v = (v - rowMean[m]) * rowRstd[m] * ln1_g[kg] + ln1_b[kg];
      int half = k >> 4;
      int k2 = k - 16*half;
      int e = k2;
      ((__bf16*)&aF[(m >> 4)*32 + (m & 15) + 16*half])[e] = f2bf(v);
    }
    // stage B: w_red panel 32k x 256n, fragment order (B: k = e + 16*half)
    {
      int k = tid >> 3, nb = tid & 7;
      const float* wr = w_red + (size_t)(K0 + k) * DRED + nb*32;
      int lhalf = k >> 4;
      int e = k & 15;
      #pragma unroll
      for (int n0 = 0; n0 < 32; ++n0) {
        int n = nb*32 + n0;
        ((__bf16*)&bF[(n >> 4)*32 + (n & 15) + 16*lhalf])[e] = f2bf(wr[n0]);
      }
    }
    __syncthreads();
    v16bf a = aF[rt*32 + lane];
    #pragma unroll
    for (int j = 0; j < 4; ++j) {
      v16bf b = bF[(cg*4 + j)*32 + lane];
      acc[j] = __builtin_amdgcn_wmma_f32_16x16x32_bf16(false, a, false, b,
                                                       (short)0, acc[j], false, false);
    }
  }
  __syncthreads();

  // ---- spill accumulators (D layout: row = r + 8*(lane>>4), col = lane&15) ----
  {
    int half = lane >> 4, nl = lane & 15;
    #pragma unroll
    for (int j = 0; j < 4; ++j) {
      int ct = cg*4 + j;
      #pragma unroll
      for (int r8 = 0; r8 < 8; ++r8)
        yBuf[(rt*16 + r8 + 8*half)*256 + ct*16 + nl] = acc[j][r8];
    }
  }
  __syncthreads();

  // ---- LN2 stats ----
  {
    int r = tid >> 3, s = tid & 7;
    float sm = 0.f, sq = 0.f;
    for (int i = 0; i < 32; ++i) {
      float v = yBuf[r*256 + s + 8*i];
      sm += v; sq += v*v;
    }
    red[tid] = sm; red2[tid] = sq;
  }
  __syncthreads();
  if (tid < 32) {
    float sm = 0.f, sq = 0.f;
    for (int j = 0; j < 8; ++j) { sm += red[tid*8+j]; sq += red2[tid*8+j]; }
    float mean = sm * (1.f/256.f);
    float var  = sq * (1.f/256.f) - mean*mean;
    rowMean[tid] = mean;
    rowRstd[tid] = rsqrtf(var + 1e-5f);
  }
  __syncthreads();
  // normalize + write yf
  for (int row = 0; row < 32; ++row) {
    int col = tid;
    float v = (yBuf[row*256+col] - rowMean[row]) * rowRstd[row] * ln2_g[col] + ln2_b[col];
    yBuf[row*256+col] = v;
    yf[(size_t)(gm0+row)*DRED + col] = v;
  }
  __syncthreads();

  // ---- gating: logits + top-2 softmax ----
  {
    int task = *taskp;
    if (tid < 128) {
      int r = tid >> 2, e = tid & 3;
      const float* wg = w_gate + (size_t)task * DRED * NEXP;
      float s = 0.f;
      for (int c = 0; c < DRED; ++c) s += yBuf[r*256+c] * wg[c*NEXP + e];
      lgs[r*4+e] = s;
    }
    __syncthreads();
    if (tid < 32) {
      int r = tid;
      float vv[4] = {lgs[r*4+0], lgs[r*4+1], lgs[r*4+2], lgs[r*4+3]};
      int i1 = 0;
      for (int e = 1; e < 4; ++e) if (vv[e] > vv[i1]) i1 = e;     // first max (jax tie rule)
      int i2 = -1;
      for (int e = 0; e < 4; ++e) {
        if (e == i1) continue;
        if (i2 < 0 || vv[e] > vv[i2]) i2 = e;
      }
      float mx = vv[i1];
      float g1 = __expf(vv[i1]-mx), g2 = __expf(vv[i2]-mx);
      float inv = 1.f/(g1+g2);
      float gv[4] = {0.f,0.f,0.f,0.f};
      gv[i1] = g1*inv; gv[i2] = g2*inv;
      for (int e = 0; e < 4; ++e) gates[(size_t)(gm0+r)*NEXP + e] = gv[e];
    }
  }
}

// ---------------------------------------------------------------------------
// k2: MoE experts (WMMA) -> per-token sigmoid pool gates gx/gt
// block = 256 thr (8 waves); each wave owns a 16-row tile
// ---------------------------------------------------------------------------
__global__ __launch_bounds__(256)
void k2_moe(const float* __restrict__ yf, const float* __restrict__ gates,
            const float* __restrict__ w1, const float* __restrict__ b1,
            const float* __restrict__ w2, const float* __restrict__ b2,
            float* __restrict__ gx, float* __restrict__ gt)
{
  __shared__ float hBuf[8][16*32];   // h transpose staging, per wave
  __shared__ float redB[8][32*16];   // lane reduction, per wave

  const int tid  = threadIdx.x;
  const int lane = tid & 31;
  const int w    = tid >> 5;
  const int gm0  = (blockIdx.x*8 + w) * 16;
  const int half = lane >> 4;
  const int nl   = lane & 15;

  // preload A fragments of yf tile (16 x 256 bf16 -> 8 k-steps)
  v16bf af[8];
  #pragma unroll
  for (int kt = 0; kt < 8; ++kt) {
    const float* yr = yf + (size_t)(gm0 + nl) * DRED;
    v16bf frag;
    #pragma unroll
    for (int e = 0; e < 16; ++e)
      frag[e] = f2bf(yr[kt*32 + a_k_of_e(e, half)]);
    af[kt] = frag;
  }

  float sx[8], st[8];
  #pragma unroll
  for (int r = 0; r < 8; ++r) { sx[r] = 0.f; st[r] = 0.f; }
  v8f zero8;
  #pragma unroll
  for (int r = 0; r < 8; ++r) zero8[r] = 0.f;

  for (int ex = 0; ex < NEXP; ++ex) {
    // h = relu(yf @ w1[ex] + b1[ex])   (16 x 32)
    v8f h0 = zero8, h1 = zero8;
    const float* w1e = w1 + (size_t)ex * DRED * DHID;
    #pragma unroll
    for (int kt = 0; kt < 8; ++kt) {
      v16bf b0f, b1f;
      #pragma unroll
      for (int e = 0; e < 16; ++e) {
        int k = kt*32 + e + 16*half;
        b0f[e] = f2bf(w1e[(size_t)k*DHID + nl]);
        b1f[e] = f2bf(w1e[(size_t)k*DHID + 16 + nl]);
      }
      h0 = __builtin_amdgcn_wmma_f32_16x16x32_bf16(false, af[kt], false, b0f, (short)0, h0, false, false);
      h1 = __builtin_amdgcn_wmma_f32_16x16x32_bf16(false, af[kt], false, b1f, (short)0, h1, false, false);
    }
    float bb0 = b1[ex*DHID + nl];
    float bb1 = b1[ex*DHID + 16 + nl];
    __syncthreads();                       // protect prior-iteration hBuf reads
    #pragma unroll
    for (int r = 0; r < 8; ++r) {
      int row = r + 8*half;
      hBuf[w][row*32 + nl]      = fmaxf(h0[r] + bb0, 0.f);
      hBuf[w][row*32 + 16 + nl] = fmaxf(h1[r] + bb1, 0.f);
    }
    __syncthreads();
    // transpose h into A layout (K=32, single WMMA step)
    v16bf a2;
    #pragma unroll
    for (int e = 0; e < 16; ++e)
      a2[e] = f2bf(hBuf[w][nl*32 + a_k_of_e(e, half)]);

    float garr[8];
    #pragma unroll
    for (int r = 0; r < 8; ++r)
      garr[r] = gates[(size_t)(gm0 + r + 8*half)*NEXP + ex];

    const float* w2e = w2 + (size_t)ex * DHID * DRED;
    const float* b2e = b2 + (size_t)ex * DRED;
    for (int ct = 0; ct < 16; ++ct) {
      int n = ct*16 + nl;
      v16bf bf2;
      #pragma unroll
      for (int e = 0; e < 16; ++e)
        bf2[e] = f2bf(w2e[(size_t)(e + 16*half)*DRED + n]);
      v8f d = __builtin_amdgcn_wmma_f32_16x16x32_bf16(false, a2, false, bf2, (short)0, zero8, false, false);
      float bias = b2e[n];
      #pragma unroll
      for (int r = 0; r < 8; ++r) {
        float contrib = garr[r] * (d[r] + bias);
        if (n < 128) sx[r] += contrib; else st[r] += contrib;
      }
    }
  }

  // reduce across lanes -> per-row means -> sigmoid gates
  #pragma unroll
  for (int r = 0; r < 8; ++r) {
    redB[w][lane*16 + r]     = sx[r];
    redB[w][lane*16 + 8 + r] = st[r];
  }
  __syncthreads();
  if (lane < 16) {
    int rr = lane;
    int hh = rr >> 3, r = rr & 7;
    float sumx = 0.f, sumt = 0.f;
    for (int l2 = 0; l2 < 16; ++l2) {
      sumx += redB[w][(hh*16 + l2)*16 + r];
      sumt += redB[w][(hh*16 + l2)*16 + 8 + r];
    }
    gx[gm0+rr] = 1.f/(1.f + __expf(-sumx*(1.f/128.f)));
    gt[gm0+rr] = 1.f/(1.f + __expf(-sumt*(1.f/128.f)));
  }
}

// ---------------------------------------------------------------------------
// k3: out_x/out_t streaming + d1/d2 mean-abs reduction (deterministic)
// grid = B*4 blocks; each thread owns one channel c, loops over N
// ---------------------------------------------------------------------------
__global__ __launch_bounds__(256)
void k3_outxt(const float* __restrict__ x, const float* __restrict__ t,
              const float* __restrict__ past_x, const float* __restrict__ past_t,
              const float* __restrict__ modal_shifts, const int* __restrict__ taskp,
              const float* __restrict__ gx, const float* __restrict__ gt,
              float* __restrict__ out_x, float* __restrict__ out_t,
              float* __restrict__ d1, float* __restrict__ d2)
{
  int b = blockIdx.x >> 2;
  int c = (blockIdx.x & 3)*256 + threadIdx.x;
  int task = *taskp;
  float shx = modal_shifts[(size_t)(task*2+0)*DIMC + c];
  float sht = modal_shifts[(size_t)(task*2+1)*DIMC + c];
  float a1 = 0.f, a2 = 0.f;
  for (int n = 0; n < BN; ++n) {
    size_t idx = ((size_t)b*BN + n)*DIMC + c;
    float ox = gx[b*BN + n] * x[idx] + shx;
    float ot = gt[b*BN + n] * t[idx] + sht;
    out_x[idx] = ox; out_t[idx] = ot;
    a1 += fabsf(ox - past_x[idx]);
    a2 += fabsf(ot - past_t[idx]);
  }
  d1[b*DIMC + c] = a1 * (1.f/BN);
  d2[b*DIMC + c] = a2 * (1.f/BN);
}

// ---------------------------------------------------------------------------
// k4: score net  s = relu(d @ ws1) @ ws2 ; per-batch softmax weights
// ---------------------------------------------------------------------------
__global__ __launch_bounds__(256)
void k4_score(const float* __restrict__ d1, const float* __restrict__ d2,
              const float* __restrict__ ws1, const float* __restrict__ ws2,
              float* __restrict__ wsm)
{
  __shared__ float dsh[DIMC];
  __shared__ float red[256];
  __shared__ float sres[2];
  int b = blockIdx.x, tid = threadIdx.x;
  for (int br = 0; br < 2; ++br) {
    const float* dv = (br == 0 ? d1 : d2) + (size_t)b*DIMC;
    for (int i = tid; i < DIMC; i += 256) dsh[i] = dv[i];
    __syncthreads();
    float s = 0.f;
    for (int jj = 0; jj < 4; ++jj) {
      int j = tid + 256*jj;
      float h = 0.f;
      for (int k = 0; k < DIMC; ++k) h += dsh[k] * ws1[(size_t)k*DIMC + j];
      s += fmaxf(h, 0.f) * ws2[j];
    }
    red[tid] = s;
    __syncthreads();
    for (int off = 128; off > 0; off >>= 1) {
      if (tid < off) red[tid] += red[tid + off];
      __syncthreads();
    }
    if (tid == 0) sres[br] = red[0];
    __syncthreads();
  }
  if (tid == 0) {
    float s1 = sres[0], s2 = sres[1];
    float m = fmaxf(s1, s2);
    float e1 = __expf(s1 - m), e2 = __expf(s2 - m);
    float inv = 1.f / (e1 + e2);
    wsm[b*2 + 0] = e1 * inv;   // w[:,0]
    wsm[b*2 + 1] = e2 * inv;   // w[:,1]
  }
}

// ---------------------------------------------------------------------------
// k_aux: importance/load reduction -> aux_loss (deterministic tree)
// ---------------------------------------------------------------------------
__global__ __launch_bounds__(256)
void k_aux(const float* __restrict__ gates, float* __restrict__ aux_out)
{
  __shared__ float rimp[256*4];
  __shared__ float rld[256*4];
  int tid = threadIdx.x;
  float imp[4] = {0,0,0,0}, ld[4] = {0,0,0,0};
  for (int i = tid; i < MTOK; i += 256) {
    #pragma unroll
    for (int e = 0; e < 4; ++e) {
      float g = gates[(size_t)i*4 + e];
      imp[e] += g;
      ld[e]  += (g > 0.f) ? 1.f : 0.f;
    }
  }
  #pragma unroll
  for (int e = 0; e < 4; ++e) { rimp[tid*4+e] = imp[e]; rld[tid*4+e] = ld[e]; }
  __syncthreads();
  for (int off = 128; off > 0; off >>= 1) {
    if (tid < off)
      for (int e = 0; e < 4; ++e) {
        rimp[tid*4+e] += rimp[(tid+off)*4+e];
        rld[tid*4+e]  += rld[(tid+off)*4+e];
      }
    __syncthreads();
  }
  if (tid == 0) {
    float aux = 0.f;
    for (int which = 0; which < 2; ++which) {
      const float* v = which ? rld : rimp;
      float mean = (v[0]+v[1]+v[2]+v[3]) * 0.25f;
      float var = 0.f;
      for (int e = 0; e < 4; ++e) { float dd = v[e]-mean; var += dd*dd; }
      var *= 0.25f;
      aux += var / (mean*mean + 1e-10f);
    }
    *aux_out = aux;
  }
}

// ---------------------------------------------------------------------------
// k5: out = w[:,1]*out_x + w[:,0]*out_t
// ---------------------------------------------------------------------------
__global__ __launch_bounds__(256)
void k5_combine(const float* __restrict__ out_x, const float* __restrict__ out_t,
                const float* __restrict__ wsm, float* __restrict__ out)
{
  size_t i = (size_t)blockIdx.x * 256 + threadIdx.x;
  int b = (int)(i >> 21);                       // N*DIM = 2^21
  out[i] = wsm[b*2+1] * out_x[i] + wsm[b*2+0] * out_t[i];
}

// ---------------------------------------------------------------------------
extern "C" void kernel_launch(void* const* d_in, const int* in_sizes, int n_in,
                              void* d_out, int out_size, void* d_ws, size_t ws_size,
                              hipStream_t stream) {
  (void)in_sizes; (void)n_in; (void)out_size; (void)ws_size;
  const float* x            = (const float*)d_in[0];
  const float* t            = (const float*)d_in[1];
  const float* past_x       = (const float*)d_in[2];
  const float* past_t       = (const float*)d_in[3];
  const int*   taskp        = (const int*)  d_in[4];
  const float* ln1_g        = (const float*)d_in[5];
  const float* ln1_b        = (const float*)d_in[6];
  const float* w_red        = (const float*)d_in[7];
  const float* ln2_g        = (const float*)d_in[8];
  const float* ln2_b        = (const float*)d_in[9];
  const float* w_gate       = (const float*)d_in[10];
  const float* w1           = (const float*)d_in[11];
  const float* b1           = (const float*)d_in[12];
  const float* w2           = (const float*)d_in[13];
  const float* b2           = (const float*)d_in[14];
  const float* modal_shifts = (const float*)d_in[15];
  const float* ws1          = (const float*)d_in[16];
  const float* ws2          = (const float*)d_in[17];

  // workspace layout (floats)
  float* wsf   = (float*)d_ws;
  float* yf    = wsf;                        // 16384*256
  float* gates = yf + (size_t)MTOK*DRED;     // 16384*4
  float* gx    = gates + (size_t)MTOK*NEXP;  // 16384
  float* gt    = gx + MTOK;                  // 16384
  float* d1    = gt + MTOK;                  // 8192
  float* d2    = d1 + BB*DIMC;               // 8192
  float* wsm   = d2 + BB*DIMC;               // 16

  // output layout: out | aux_loss | out_x | out_t
  float* out   = (float*)d_out;
  float* auxp  = out + (size_t)BB*BN*DIMC;
  float* out_x = auxp + 1;
  float* out_t = out_x + (size_t)BB*BN*DIMC;

  k1_ln_reduce<<<MTOK/32, 256, 0, stream>>>(x, t, ln1_g, ln1_b, w_red,
                                            ln2_g, ln2_b, w_gate, taskp, yf, gates);
  k2_moe<<<MTOK/128, 256, 0, stream>>>(yf, gates, w1, b1, w2, b2, gx, gt);
  k3_outxt<<<BB*4, 256, 0, stream>>>(x, t, past_x, past_t, modal_shifts, taskp,
                                     gx, gt, out_x, out_t, d1, d2);
  k4_score<<<BB, 256, 0, stream>>>(d1, d2, ws1, ws2, wsm);
  k_aux<<<1, 256, 0, stream>>>(gates, auxp);
  k5_combine<<<(BB*BN*DIMC)/256, 256, 0, stream>>>(out_x, out_t, wsm, out);
}
